// Scaled_DotProduct_77257871720516
// MI455X (gfx1250) — matrix operand order repaired and checked by reading, not compiled, not run on hardware
//
#include <hip/hip_runtime.h>

typedef __attribute__((ext_vector_type(16))) _Float16 v16h;
typedef __attribute__((ext_vector_type(8)))  float    v8f;
typedef __attribute__((ext_vector_type(4)))  float    f32x4;

#define B_  8
#define H_  12
#define S_  1024
#define D_  64
#define QT  32                 // query rows per workgroup
#define LDST 1036              // padded LDS row stride (floats): %64==12, %4==0
#define NTHREADS 256

// A-fragment (16x32 f16, M x K): lane holds row M=lane%16.
//   e in [0,8):  k = kbase + 16*lhalf + e
//   e in [8,16): k = kbase + 16*lhalf + e
// rowp points at (row, kbase). mul applied before f16 convert.
__device__ __forceinline__ v16h make_a_frag(const float* rowp, int lhalf, float mul) {
    const float* p0 = rowp + 16 * lhalf;
    const float* p1 = rowp + 16 * lhalf + 8;
    float t[16];
    *(float4*)(t + 0)  = *(const float4*)(p0);
    *(float4*)(t + 4)  = *(const float4*)(p0 + 4);
    *(float4*)(t + 8)  = *(const float4*)(p1);
    *(float4*)(t + 12) = *(const float4*)(p1 + 4);
    v16h f;
#pragma unroll
    for (int e = 0; e < 16; ++e) f[e] = (_Float16)(t[e] * mul);
    return f;
}

// B-fragment (32x16 f16, K x N) where the lane's 16 k-values are contiguous
// in memory: p points at the first of 16 consecutive floats.
__device__ __forceinline__ v16h make_b_frag_contig(const float* p) {
    float t[16];
    *(float4*)(t + 0)  = *(const float4*)(p);
    *(float4*)(t + 4)  = *(const float4*)(p + 4);
    *(float4*)(t + 8)  = *(const float4*)(p + 8);
    *(float4*)(t + 12) = *(const float4*)(p + 12);
    v16h f;
#pragma unroll
    for (int e = 0; e < 16; ++e) f[e] = (_Float16)t[e];
    return f;
}

__global__ __launch_bounds__(NTHREADS)
void attn_fused_kernel(const float* __restrict__ q,
                       const float* __restrict__ k,
                       const float* __restrict__ v,
                       const int*   __restrict__ scale_p,
                       float* __restrict__ out)
{
    extern __shared__ float lds[];          // QT x LDST floats + QT floats (rs)
    float* lds_rs = lds + QT * LDST;

    const int tid   = threadIdx.x;
    const int wave  = tid >> 5;             // 0..7
    const int lane  = tid & 31;
    const int lhalf = lane >> 4;            // 0 or 1
    const int l16   = lane & 15;

    const int bh    = blockIdx.x >> 5;      // 0..95  (B*H)
    const int qt    = blockIdx.x & 31;      // 0..31  (S/QT)
    const int qbase = qt * QT;

    const float inv_scale = 1.0f / (float)scale_p[0];

    const float* Qh = q + (size_t)bh * S_ * D_;
    const float* Kh = k + (size_t)bh * S_ * D_;
    const float* Vh = v + (size_t)bh * S_ * D_;
    float* Pout = out + (size_t)bh * S_ * S_;
    float* Oout = out + (size_t)B_ * H_ * S_ * S_ + (size_t)bh * S_ * D_;

    // ---------------- Phase 1: S = (Q/scale) K^T into LDS (WMMA) ----------
    {
        const int mt    = wave & 1;         // m-tile within the 32-row block
        const int wquad = wave >> 1;        // 0..3 : n-tile phase

        const float* qrow = Qh + (size_t)(qbase + 16 * mt + l16) * D_;
        const v16h a0 = make_a_frag(qrow,      lhalf, inv_scale);  // d = 0..31
        const v16h a1 = make_a_frag(qrow + 32, lhalf, inv_scale);  // d = 32..63

        for (int j = 0; j < 16; ++j) {
            const int nbase = (wquad + 4 * j) * 16;   // key tile base
            // B: lane -> key row nbase+l16, d = dbase + 16*lhalf + [0..15]
            const float* krow = Kh + (size_t)(nbase + l16) * D_ + 16 * lhalf;
            const v16h b0 = make_b_frag_contig(krow);        // d = 0..31
            const v16h b1 = make_b_frag_contig(krow + 32);   // d = 32..63

            v8f c = {0.f, 0.f, 0.f, 0.f, 0.f, 0.f, 0.f, 0.f};
            c = __builtin_amdgcn_wmma_f32_16x16x32_f16(false, a0, false, b0,
                                                       (short)0, c, false, false);
            c = __builtin_amdgcn_wmma_f32_16x16x32_f16(false, a1, false, b1,
                                                       (short)0, c, false, false);
#pragma unroll
            for (int r = 0; r < 8; ++r) {
                const int m = 16 * mt + r + 8 * lhalf;      // LDS row 0..31
                lds[m * LDST + nbase + l16] = c[r];
            }
        }
    }
    __syncthreads();

    // ---------------- Phase 2a: row max + exp + sum (8 threads / row) -----
    {
        const int row = tid >> 3;           // 0..31
        const int jj  = tid & 7;
        float* rowp = lds + row * LDST;

        float mx = -__builtin_inff();
#pragma unroll 8
        for (int i = 0; i < 128; ++i) mx = fmaxf(mx, rowp[jj + 8 * i]);
        mx = fmaxf(mx, __shfl_xor(mx, 1, 32));
        mx = fmaxf(mx, __shfl_xor(mx, 2, 32));
        mx = fmaxf(mx, __shfl_xor(mx, 4, 32));

        float sum = 0.f;
#pragma unroll 8
        for (int i = 0; i < 128; ++i) {
            const int n = jj + 8 * i;
            const float e = __expf(rowp[n] - mx);
            rowp[n] = e;
            sum += e;
        }
        sum += __shfl_xor(sum, 1, 32);
        sum += __shfl_xor(sum, 2, 32);
        sum += __shfl_xor(sum, 4, 32);
        if (jj == 0) lds_rs[row] = 1.0f / sum;
    }
    __syncthreads();

    // ---------------- Phase 2b: normalize + stream P (coalesced b128 NT) --
    {
#pragma unroll 4
        for (int i = 0; i < 32; ++i) {
            const int f   = i * 256 + tid;      // float4 chunk id 0..8191
            const int row = f >> 8;             // 256 chunks per row
            const int col = (f & 255) * 4;
            const float rsv = lds_rs[row];
            f32x4 x = *(const f32x4*)(lds + row * LDST + col);
            x *= rsv;
            *(f32x4*)(lds + row * LDST + col) = x;   // normalized P for PV
            __builtin_nontemporal_store(
                x, (f32x4*)(Pout + (size_t)(qbase + row) * S_ + col));
        }
    }
    __syncthreads();

    // ---------------- Phase 3: O = P @ V (WMMA), one 16x16 tile per wave --
    {
        const int mt2 = wave >> 2;          // 0/1
        const int nt2 = wave & 3;           // 0..3  (D/16)
        const int d   = 16 * nt2 + l16;     // output feature for this lane (B frag)

        v8f o = {0.f, 0.f, 0.f, 0.f, 0.f, 0.f, 0.f, 0.f};
        const float* prow = lds + (size_t)(16 * mt2 + l16) * LDST;

        for (int kb = 0; kb < S_; kb += 32) {
            const v16h pa = make_a_frag(prow + kb, lhalf, 1.0f);

            v16h vb;
            const int key0 = kb + 16 * lhalf;
            const float* vp = Vh + (size_t)key0 * D_ + d;
#pragma unroll
            for (int e = 0; e < 16; ++e)
                vb[e] = (_Float16)vp[(size_t)e * D_];   // lanes 0..15 coalesce per e

            o = __builtin_amdgcn_wmma_f32_16x16x32_f16(false, pa, false, vb,
                                                       (short)0, o, false, false);
        }
#pragma unroll
        for (int r = 0; r < 8; ++r) {
            const int m = 16 * mt2 + r + 8 * lhalf;
            __builtin_nontemporal_store(
                o[r], Oout + (size_t)(qbase + m) * D_ + d);
        }
    }
}

extern "C" void kernel_launch(void* const* d_in, const int* in_sizes, int n_in,
                              void* d_out, int out_size, void* d_ws, size_t ws_size,
                              hipStream_t stream) {
    (void)in_sizes; (void)n_in; (void)out_size; (void)d_ws; (void)ws_size;
    const float* q = (const float*)d_in[0];
    const float* k = (const float*)d_in[1];
    const float* v = (const float*)d_in[2];
    const int* scale = (const int*)d_in[3];
    float* out = (float*)d_out;

    const int nblocks = B_ * H_ * (S_ / QT);                    // 3072
    const size_t smem = ((size_t)QT * LDST + QT) * sizeof(float);  // 132,736 B
    attn_fused_kernel<<<nblocks, NTHREADS, smem, stream>>>(q, k, v, scale, out);
}
